// PerformerEncoderLayer_70025146794703
// MI455X (gfx1250) — matrix operand order repaired and checked by reading, not compiled, not run on hardware
//
#include <hip/hip_runtime.h>
#include <cstdint>
#include <cstddef>

// ---------------------------------------------------------------------------
// Problem constants (PerformerEncoderLayer)
// ---------------------------------------------------------------------------
#define B_    4
#define L_    4096
#define D_    1024
#define H_    16
#define DFF_  4096
#define M_    256
#define DH_   64
#define BL_   (B_ * L_)          // 16384 token rows

// dn = DH^-0.25, 0.5*dn^2 = 0.0625, ratio = 1/sqrt(M) = 0.0625
#define DN_F       0.35355339059327373f
#define HALF_DN2_F 0.0625f
#define RATIO_F    0.0625f
#define EPS_K_F    1e-6f
#define EPS_LN_F   1e-6f

typedef __bf16 bf16;
typedef __attribute__((ext_vector_type(8)))  bf16  bf16x8;
typedef __attribute__((ext_vector_type(16))) bf16  bf16x16;
typedef __attribute__((ext_vector_type(8)))  float f32x8;

// ---------------------------------------------------------------------------
// Tiled bf16 WMMA GEMM, TN layout:
//   C[M x N] (f32) = A[M x K] (row-major bf16)  *  Bt[N x K]^T (row-major bf16)
// Block tile 128 x BN, BK = 32. 8 waves:
//   BN=128 -> 2x4 wave grid, 64x32 per wave; BN=64 -> 4x2, 32x32 per wave.
// Global->LDS staging via CDNA5 async copies (GLOBAL_LOAD_ASYNC_TO_LDS_B128,
// ASYNCcnt) with LDS double buffering; the steady-state loop is software-
// pipelined (issue tile k+1, wait for tile k with s_wait_asynccnt ACNT since
// async copies retire in order, consume tile k) with a peeled final tile so
// the hot loop is branch-free.
// Z batching: A/B/C offset by (z/zdiv)*O + (z%zdiv)*I elements.
// ---------------------------------------------------------------------------
template <int BN>
__global__ __launch_bounds__(256)
void gemm_bf16_wmma(const bf16* __restrict__ Ag, const bf16* __restrict__ Btg,
                    float* __restrict__ Cg,
                    int K, int lda, int ldb, int ldc, int zdiv,
                    long long aO, long long aI, long long bO, long long bI,
                    long long cO, long long cI)
{
    constexpr int BM    = 128;
    constexpr int BK    = 32;
    constexpr int WROWS = (BN == 128) ? 2 : 4;
    constexpr int WCOLS = 8 / WROWS;
    constexpr int WTM   = BM / WROWS;     // 64 or 32
    constexpr int WTN   = BN / WCOLS;     // 32
    constexpr int TM    = WTM / 16;       // 4 or 2
    constexpr int TN    = WTN / 16;       // 2
    constexpr int LDS_S = BK + 8;         // 80B row stride: 16B aligned, bank-spread
    constexpr int BITER = BN / 64;        // B-tile async ops per thread (2 or 1)
    constexpr int ACNT  = 2 + BITER;      // async ops issued per wave per tile

    __shared__ bf16 As[2][BM][LDS_S];
    __shared__ bf16 Bs[2][BN][LDS_S];

    const int tid  = threadIdx.x;
    const int lane = tid & 31;
    const int wave = tid >> 5;
    const int wrow = wave % WROWS;
    const int wcol = wave / WROWS;
    const int half = lane >> 4;           // wave32 lane-half selects K range
    const int lr   = lane & 15;

    const int z  = blockIdx.z;
    const int zo = z / zdiv;
    const int zi = z % zdiv;
    const bf16* A  = Ag  + (size_t)zo * (size_t)aO + (size_t)zi * (size_t)aI;
    const bf16* Bt = Btg + (size_t)zo * (size_t)bO + (size_t)zi * (size_t)bI;
    float*      C  = Cg  + (size_t)zo * (size_t)cO + (size_t)zi * (size_t)cI;

    const int rowBase = blockIdx.x * BM;
    const int colBase = blockIdx.y * BN;

    // Issue one K-tile's async copies (per-lane 16B chunks, LDS-scatter keeps
    // the padded LDS stride).  ASYNCcnt += ACNT per wave.
    auto issue = [&](int k0, int buf) {
        #pragma unroll
        for (int it = 0; it < 2; ++it) {              // A: 512 chunks of 8 bf16
            int c  = tid + it * 256;
            int ar = c >> 2;                           // 4 chunks per row
            int kc = (c & 3) * 8;
            const bf16* src = A + (size_t)(rowBase + ar) * lda + (k0 + kc);
            unsigned dst = (unsigned)(size_t)&As[buf][ar][kc];
            asm volatile("global_load_async_to_lds_b128 %0, %1, off"
                         :: "v"(dst), "v"(src) : "memory");
        }
        #pragma unroll
        for (int it = 0; it < BITER; ++it) {          // Bt rows are K-contiguous
            int c  = tid + it * 256;
            int nr = c >> 2;
            int kc = (c & 3) * 8;
            const bf16* src = Bt + (size_t)(colBase + nr) * ldb + (k0 + kc);
            unsigned dst = (unsigned)(size_t)&Bs[buf][nr][kc];
            asm volatile("global_load_async_to_lds_b128 %0, %1, off"
                         :: "v"(dst), "v"(src) : "memory");
        }
    };

    f32x8 acc[TM][TN] = {};

    // Consume one staged K-tile: fragment loads per ISA 7.12.2 + 8/4 WMMAs.
    auto compute = [&](int cur) {
        // A 16x32: lane(h,r) holds k = h*8+[0..7] and h*8+16+[0..7] of row r
        bf16x16 afrag[TM];
        #pragma unroll
        for (int i = 0; i < TM; ++i) {
            int r  = wrow * WTM + i * 16 + lr;
            int ka = half * 8;
            bf16x8 lo = *(const bf16x8*)&As[cur][r][ka];
            bf16x8 hi = *(const bf16x8*)&As[cur][r][ka + 16];
            #pragma unroll
            for (int t = 0; t < 8; ++t) { afrag[i][t] = lo[t]; afrag[i][t + 8] = hi[t]; }
        }
        // B 32x16: lane(h,n) holds k = h*8+[0..7] and h*8+16+[0..7] of column n
        bf16x16 bfrag[TN];
        #pragma unroll
        for (int j = 0; j < TN; ++j) {
            int n  = wcol * WTN + j * 16 + lr;
            int kb = half * 8;
            bf16x8 lo = *(const bf16x8*)&Bs[cur][n][kb];
            bf16x8 hi = *(const bf16x8*)&Bs[cur][n][kb + 16];
            #pragma unroll
            for (int t = 0; t < 8; ++t) { bfrag[j][t] = lo[t]; bfrag[j][t + 8] = hi[t]; }
        }
        #pragma unroll
        for (int i = 0; i < TM; ++i)
            #pragma unroll
            for (int j = 0; j < TN; ++j)
                acc[i][j] = __builtin_amdgcn_wmma_f32_16x16x32_bf16(
                    false, afrag[i], false, bfrag[j], (short)0, acc[i][j], false, false);
    };

    const int nk = K / BK;
    issue(0, 0);

    int ki = 0;
    for (; ki < nk - 1; ++ki) {                       // branch-free steady state
        const int cur = ki & 1;
        issue((ki + 1) * BK, cur ^ 1);                // stream next tile
        asm volatile("s_wait_asynccnt %0" :: "i"(ACNT) : "memory"); // tile ki landed
        __syncthreads();
        compute(cur);
        __syncthreads();                              // buffer reuse guard
    }
    // peeled final tile
    asm volatile("s_wait_asynccnt 0" ::: "memory");
    __syncthreads();
    compute(ki & 1);

    // ---- epilogue: C/D 16x16 f32 -> VGPR g holds row g + 8*half, col lr ----
    #pragma unroll
    for (int i = 0; i < TM; ++i)
        #pragma unroll
        for (int j = 0; j < TN; ++j) {
            int r0 = rowBase + wrow * WTM + i * 16 + half * 8;
            int c0 = colBase + wcol * WTN + j * 16 + lr;
            #pragma unroll
            for (int g = 0; g < 8; ++g)
                C[(size_t)(r0 + g) * ldc + c0] = acc[i][j][g];
        }
}

// ---------------------------------------------------------------------------
// Elementwise / reduction / transpose kernels
// ---------------------------------------------------------------------------
__global__ void k_f32_to_bf16(const float* __restrict__ in, bf16* __restrict__ out, long long n)
{
    long long i = (long long)blockIdx.x * 256 + threadIdx.x;
    if (i < n) out[i] = (bf16)in[i];
}

__global__ void k_scale_bf16(const float* __restrict__ in, bf16* __restrict__ out,
                             long long n, float s)
{
    long long i = (long long)blockIdx.x * 256 + threadIdx.x;
    if (i < n) out[i] = (bf16)(in[i] * s);
}

// f32 [K][N] row-major -> bf16 [N][K] row-major (weight pre-transpose)
__global__ void k_convT(const float* __restrict__ in, bf16* __restrict__ out, int K, int N)
{
    long long i = (long long)blockIdx.x * 256 + threadIdx.x;   // i = n*K + k
    if (i < (long long)K * N) {
        int n = (int)(i / K), k = (int)(i % K);
        out[i] = (bf16)in[(size_t)k * N + n];
    }
}

// v [b*L+l][h*DH+d] f32 -> vT[((b*H+h)*DH+d)][l] bf16
__global__ void k_vT(const float* __restrict__ v, bf16* __restrict__ out)
{
    long long i = (long long)blockIdx.x * 256 + threadIdx.x;   // B*H*DH*L total
    if (i < (long long)B_ * H_ * DH_ * L_) {
        int l  = (int)(i & (L_ - 1));
        long long r = i >> 12;                                 // (b*H+h)*DH + d
        int d  = (int)(r & (DH_ - 1));
        int bh = (int)(r >> 6);
        int b  = bh >> 4, h = bh & 15;
        out[i] = (bf16)v[(size_t)(b * L_ + l) * D_ + h * DH_ + d];
    }
}

// kv[(b*H+h)*M+m][d] f32 -> kvT[((b*H+h)*DH+d)][m] bf16
__global__ void k_kvT(const float* __restrict__ kv, bf16* __restrict__ out)
{
    int i = blockIdx.x * 256 + threadIdx.x;                    // B*H*DH*M total
    if (i < B_ * H_ * DH_ * M_) {
        int m  = i & (M_ - 1);
        int r  = i >> 8;
        int d  = r & (DH_ - 1);
        int bh = r >> 6;
        out[i] = (bf16)kv[(size_t)(bh * M_ + m) * DH_ + d];
    }
}

// diag[r][h] = 0.5*dn^2 * sum_dh q[r][h*64+dh]^2
__global__ void k_diag(const float* __restrict__ q, float* __restrict__ dg)
{
    int i = blockIdx.x * 256 + threadIdx.x;                    // (r,h)
    if (i < BL_ * H_) {
        int r = i >> 4, h = i & 15;
        const float* p = q + (size_t)r * D_ + h * DH_;
        float s = 0.f;
        for (int t = 0; t < DH_; ++t) { float v = p[t]; s += v * v; }
        dg[i] = HALF_DN2_F * s;
    }
}

__global__ void k_rowmax(const float* __restrict__ dd, float* __restrict__ mx)
{
    int i = blockIdx.x * 256 + threadIdx.x;                    // (r,h)
    if (i < BL_ * H_) {
        int r = i >> 4, h = i & 15;
        const float* p = dd + (size_t)r * (H_ * M_) + h * M_;
        float m = p[0];
        for (int t = 1; t < M_; ++t) m = fmaxf(m, p[t]);
        mx[i] = m;
    }
}

__device__ __forceinline__ unsigned f2key(float f)
{
    unsigned u = __float_as_uint(f);
    return (u & 0x80000000u) ? ~u : (u | 0x80000000u);
}
__device__ __forceinline__ float key2f(unsigned k)
{
    return (k & 0x80000000u) ? __uint_as_float(k ^ 0x80000000u) : __uint_as_float(~k);
}

__global__ void k_zero_u32(unsigned* p)
{
    if (blockIdx.x == 0 && threadIdx.x == 0) *p = 0u;
}

__global__ void k_gmax(const float* __restrict__ dd, unsigned* __restrict__ g, long long n)
{
    __shared__ unsigned red[256];
    unsigned best = 0u;
    for (long long i = (long long)blockIdx.x * 256 + threadIdx.x; i < n;
         i += (long long)gridDim.x * 256)
        best = max(best, f2key(dd[i]));
    red[threadIdx.x] = best;
    __syncthreads();
    for (int o = 128; o; o >>= 1) {
        if (threadIdx.x < o) red[threadIdx.x] = max(red[threadIdx.x], red[threadIdx.x + o]);
        __syncthreads();
    }
    if (threadIdx.x == 0) atomicMax(g, red[0]);
}

// qp[r][h*M+m] = ratio*(exp(dd - diag - rowmax) + eps) bf16, layout unchanged
__global__ void k_featq(const float* __restrict__ dd, const float* __restrict__ dg,
                        const float* __restrict__ mx, bf16* __restrict__ out, long long n)
{
    long long i = (long long)blockIdx.x * 256 + threadIdx.x;
    if (i < n) {
        int r = (int)(i >> 12);
        int h = ((int)i >> 8) & 15;
        out[i] = (bf16)(RATIO_F * (expf(dd[i] - dg[r * 16 + h] - mx[r * 16 + h]) + EPS_K_F));
    }
}

// kp written TRANSPOSED: kpT[((b*H+h)*M+m)][l] = ratio*(exp(..) + eps)
// (makes kpT a row-major [M x L] A-operand per (b,h), and ksum contiguous)
__global__ void k_featk(const float* __restrict__ dd, const float* __restrict__ dg,
                        const unsigned* __restrict__ gk, bf16* __restrict__ out, long long n)
{
    long long i = (long long)blockIdx.x * 256 + threadIdx.x;
    if (i < n) {
        float mx = key2f(*gk);
        int l  = (int)(i & (L_ - 1));
        long long r = i >> 12;                                 // (b*H+h)*M + m
        int m  = (int)(r & (M_ - 1));
        int bh = (int)(r >> 8);
        int b  = bh >> 4, h = bh & 15;
        size_t src = (size_t)(b * L_ + l) * (H_ * M_) + h * M_ + m;
        int    row = b * L_ + l;
        out[i] = (bf16)(RATIO_F * (expf(dd[src] - dg[row * 16 + h] - mx) + EPS_K_F));
    }
}

// ksum[(b*H+h)*M+m] = sum_l kpT[..][l]  (contiguous reduction)
__global__ void k_ksum(const bf16* __restrict__ kpT, float* __restrict__ ks)
{
    int i = blockIdx.x * 256 + threadIdx.x;
    if (i < B_ * H_ * M_) {
        const bf16* p = kpT + (size_t)i * L_;
        float s = 0.f;
        for (int l = 0; l < L_; ++l) s += (float)p[l];
        ks[i] = s;
    }
}

// denom[r][h] = sum_m qp[r][h*M+m] * ksum[b,h,m]
__global__ void k_denom(const bf16* __restrict__ qp, const float* __restrict__ ks,
                        float* __restrict__ dn)
{
    int i = blockIdx.x * 256 + threadIdx.x;                    // (r,h)
    if (i < BL_ * H_) {
        int r = i >> 4, h = i & 15, b = r >> 12;
        const bf16*  p = qp + (size_t)r * (H_ * M_) + h * M_;
        const float* s = ks + (size_t)(b * H_ + h) * M_;
        float acc = 0.f;
        for (int m = 0; m < M_; ++m) acc += (float)p[m] * s[m];
        dn[i] = acc;
    }
}

// attn_bf16 = num / denom (broadcast over DH)
__global__ void k_attn(const float* __restrict__ num, const float* __restrict__ den,
                       bf16* __restrict__ out, long long n)
{
    long long i = (long long)blockIdx.x * 256 + threadIdx.x;
    if (i < n) {
        int r = (int)(i >> 10);
        int h = ((int)i >> 6) & 15;
        out[i] = (bf16)(num[i] / den[r * 16 + h]);
    }
}

// h = elu(ffn1 + b1) as bf16
__global__ void k_bias_elu(const float* __restrict__ in, const float* __restrict__ b,
                           bf16* __restrict__ out, long long n)
{
    long long i = (long long)blockIdx.x * 256 + threadIdx.x;
    if (i < n) {
        float v = in[i] + b[(int)(i & (DFF_ - 1))];
        out[i] = (bf16)(v > 0.f ? v : expm1f(v));
    }
}

// out = LayerNorm(x + y [+ bias]) ; optional f32 and bf16 outputs
__global__ __launch_bounds__(256)
void k_res_ln(const float* __restrict__ x, const float* __restrict__ y,
              const float* __restrict__ bias,
              const float* __restrict__ g, const float* __restrict__ be,
              float* __restrict__ of, bf16* __restrict__ ob)
{
    __shared__ float buf[D_];
    __shared__ float r1[256], r2[256];
    const int row = blockIdx.x, tid = threadIdx.x;
    const size_t base = (size_t)row * D_;
    float s = 0.f, s2 = 0.f;
    for (int d = tid; d < D_; d += 256) {
        float v = x[base + d] + y[base + d] + (bias ? bias[d] : 0.f);
        buf[d] = v; s += v; s2 += v * v;
    }
    r1[tid] = s; r2[tid] = s2;
    __syncthreads();
    for (int o = 128; o; o >>= 1) {
        if (tid < o) { r1[tid] += r1[tid + o]; r2[tid] += r2[tid + o]; }
        __syncthreads();
    }
    float mean = r1[0] * (1.f / D_);
    float var  = r2[0] * (1.f / D_) - mean * mean;
    float inv  = rsqrtf(var + EPS_LN_F);
    for (int d = tid; d < D_; d += 256) {
        float o = (buf[d] - mean) * inv * g[d] + be[d];
        if (of) of[base + d] = o;
        if (ob) ob[base + d] = (bf16)o;
    }
}

// ---------------------------------------------------------------------------
// Host driver
// ---------------------------------------------------------------------------
#define EW(n) dim3((unsigned)((((long long)(n)) + 255) / 256))

extern "C" void kernel_launch(void* const* d_in, const int* in_sizes, int n_in,
                              void* d_out, int out_size, void* d_ws, size_t ws_size,
                              hipStream_t stream)
{
    (void)in_sizes; (void)n_in; (void)out_size; (void)ws_size;

    const float* x    = (const float*)d_in[0];
    const float* wq   = (const float*)d_in[1];
    const float* wk   = (const float*)d_in[2];
    const float* wv   = (const float*)d_in[3];
    const float* wo   = (const float*)d_in[4];
    const float* proj = (const float*)d_in[5];
    const float* ln1g = (const float*)d_in[6];
    const float* ln1b = (const float*)d_in[7];
    const float* ln2g = (const float*)d_in[8];
    const float* ln2b = (const float*)d_in[9];
    const float* w1   = (const float*)d_in[10];
    const float* b1   = (const float*)d_in[11];
    const float* w2   = (const float*)d_in[12];
    const float* b2   = (const float*)d_in[13];
    float* out = (float*)d_out;

    // ---- workspace bump allocator (256B aligned) ----
    char* p = (char*)d_ws;
    auto alloc = [&](size_t bytes) -> void* {
        void* r = (void*)p;
        p += (bytes + 255) & ~(size_t)255;
        return r;
    };
    bf16*  xb    = (bf16*) alloc((size_t)BL_ * D_ * 2);
    bf16*  wqT   = (bf16*) alloc((size_t)D_ * D_ * 2);       // [N][K] bf16
    bf16*  wkT   = (bf16*) alloc((size_t)D_ * D_ * 2);
    bf16*  wvT   = (bf16*) alloc((size_t)D_ * D_ * 2);
    bf16*  woT   = (bf16*) alloc((size_t)D_ * D_ * 2);
    bf16*  w1T   = (bf16*) alloc((size_t)D_ * DFF_ * 2);
    bf16*  w2T   = (bf16*) alloc((size_t)DFF_ * D_ * 2);
    bf16*  ptb   = (bf16*) alloc((size_t)M_ * DH_ * 2);      // proj*dn, already [N=M][K=DH]
    float* qf    = (float*)alloc((size_t)BL_ * D_ * 4);
    float* kf    = (float*)alloc((size_t)BL_ * D_ * 4);
    float* vf    = (float*)alloc((size_t)BL_ * D_ * 4);
    bf16*  qb    = (bf16*) alloc((size_t)BL_ * D_ * 2);
    bf16*  kb    = (bf16*) alloc((size_t)BL_ * D_ * 2);
    bf16*  vbT   = (bf16*) alloc((size_t)BL_ * D_ * 2);      // [(b h d)][l]
    float* ddq   = (float*)alloc((size_t)BL_ * H_ * M_ * 4);
    float* ddk   = (float*)alloc((size_t)BL_ * H_ * M_ * 4);
    bf16*  qpb   = (bf16*) alloc((size_t)BL_ * H_ * M_ * 2);
    bf16*  kpT   = (bf16*) alloc((size_t)BL_ * H_ * M_ * 2); // [(b h m)][l]
    float* diagq = (float*)alloc((size_t)BL_ * H_ * 4);
    float* diagk = (float*)alloc((size_t)BL_ * H_ * 4);
    float* mxq   = (float*)alloc((size_t)BL_ * H_ * 4);
    unsigned* gmax = (unsigned*)alloc(256);
    float* kvf   = (float*)alloc((size_t)B_ * H_ * M_ * DH_ * 4);
    bf16*  kvT   = (bf16*) alloc((size_t)B_ * H_ * M_ * DH_ * 2); // [(b h d)][m]
    float* ksum  = (float*)alloc((size_t)B_ * H_ * M_ * 4);
    float* denom = (float*)alloc((size_t)BL_ * H_ * 4);
    bf16*  attnb = (bf16*) alloc((size_t)BL_ * D_ * 2);
    bf16*  out1b = (bf16*) alloc((size_t)BL_ * D_ * 2);
    float* c7    = (float*)alloc((size_t)BL_ * D_ * 4);
    // Buffer reuse (producer strictly after last read of the alias):
    float* num  = qf;            // qf dead after diag/convert
    float* c5   = kf;            // attn_out
    float* out1 = vf;            // residual-1 output (f32)
    float* c6   = ddq;           // FFN hidden pre-activation
    bf16*  hb   = (bf16*)ddk;    // FFN hidden bf16

    // ---- 1. weight pre-transposes + input conversion ----
    k_convT<<<EW((long long)D_ * D_), 256, 0, stream>>>(wq, wqT, D_, D_);
    k_convT<<<EW((long long)D_ * D_), 256, 0, stream>>>(wk, wkT, D_, D_);
    k_convT<<<EW((long long)D_ * D_), 256, 0, stream>>>(wv, wvT, D_, D_);
    k_convT<<<EW((long long)D_ * D_), 256, 0, stream>>>(wo, woT, D_, D_);
    k_convT<<<EW((long long)D_ * DFF_), 256, 0, stream>>>(w1, w1T, D_, DFF_);
    k_convT<<<EW((long long)DFF_ * D_), 256, 0, stream>>>(w2, w2T, DFF_, D_);
    k_f32_to_bf16<<<EW((long long)BL_ * D_), 256, 0, stream>>>(x, xb, (long long)BL_ * D_);
    k_scale_bf16<<<EW((long long)M_ * DH_), 256, 0, stream>>>(proj, ptb,
                                                              (long long)M_ * DH_, DN_F);

    // ---- 2. QKV projections: [16384,1024] x [1024,1024]^T ----
    gemm_bf16_wmma<128><<<dim3(BL_ / 128, D_ / 128, 1), 256, 0, stream>>>(
        xb, wqT, qf, D_, D_, D_, D_, 1, 0, 0, 0, 0, 0, 0);
    gemm_bf16_wmma<128><<<dim3(BL_ / 128, D_ / 128, 1), 256, 0, stream>>>(
        xb, wkT, kf, D_, D_, D_, D_, 1, 0, 0, 0, 0, 0, 0);
    gemm_bf16_wmma<128><<<dim3(BL_ / 128, D_ / 128, 1), 256, 0, stream>>>(
        xb, wvT, vf, D_, D_, D_, D_, 1, 0, 0, 0, 0, 0, 0);

    k_f32_to_bf16<<<EW((long long)BL_ * D_), 256, 0, stream>>>(qf, qb, (long long)BL_ * D_);
    k_f32_to_bf16<<<EW((long long)BL_ * D_), 256, 0, stream>>>(kf, kb, (long long)BL_ * D_);
    k_vT<<<EW((long long)BL_ * D_), 256, 0, stream>>>(vf, vbT);
    k_diag<<<EW(BL_ * H_), 256, 0, stream>>>(qf, diagq);
    k_diag<<<EW(BL_ * H_), 256, 0, stream>>>(kf, diagk);

    // ---- 3. feature-map projections per head: [16384,64] x [256,64]^T ----
    gemm_bf16_wmma<128><<<dim3(BL_ / 128, M_ / 128, H_), 256, 0, stream>>>(
        qb, ptb, ddq, DH_, D_, DH_, H_ * M_, 1,
        (long long)DH_, 0, 0, 0, (long long)M_, 0);
    gemm_bf16_wmma<128><<<dim3(BL_ / 128, M_ / 128, H_), 256, 0, stream>>>(
        kb, ptb, ddk, DH_, D_, DH_, H_ * M_, 1,
        (long long)DH_, 0, 0, 0, (long long)M_, 0);

    // ---- 4. stabilizers + exp feature maps ----
    k_rowmax<<<EW(BL_ * H_), 256, 0, stream>>>(ddq, mxq);
    k_zero_u32<<<1, 64, 0, stream>>>(gmax);
    k_gmax<<<4096, 256, 0, stream>>>(ddk, gmax, (long long)BL_ * H_ * M_);
    k_featq<<<EW((long long)BL_ * H_ * M_), 256, 0, stream>>>(
        ddq, diagq, mxq, qpb, (long long)BL_ * H_ * M_);
    k_featk<<<EW((long long)BL_ * H_ * M_), 256, 0, stream>>>(
        ddk, diagk, gmax, kpT, (long long)BL_ * H_ * M_);
    k_ksum<<<EW(B_ * H_ * M_), 256, 0, stream>>>(kpT, ksum);

    // ---- 5. kv[b,h] = kpT[256,4096] x vT[64,4096]^T, z = b*H+h ----
    gemm_bf16_wmma<64><<<dim3(M_ / 128, 1, B_ * H_), 256, 0, stream>>>(
        kpT, vbT, kvf, L_, L_, L_, DH_, H_,
        (long long)H_ * M_ * L_,  (long long)M_ * L_,
        (long long)H_ * DH_ * L_, (long long)DH_ * L_,
        (long long)H_ * M_ * DH_, (long long)M_ * DH_);
    k_kvT<<<EW((long long)B_ * H_ * M_ * DH_), 256, 0, stream>>>(kvf, kvT);

    // ---- 6. num[b,:,h,:] = qp[4096,256] x kvT[64,256]^T, z = b*H+h ----
    gemm_bf16_wmma<64><<<dim3(L_ / 128, 1, B_ * H_), 256, 0, stream>>>(
        qpb, kvT, num, M_, H_ * M_, M_, D_, H_,
        (long long)L_ * H_ * M_,  (long long)M_,
        (long long)H_ * DH_ * M_, (long long)DH_ * M_,
        (long long)L_ * D_,       (long long)DH_);

    // ---- 7. denom + attn, output projection ----
    k_denom<<<EW(BL_ * H_), 256, 0, stream>>>(qpb, ksum, denom);
    k_attn<<<EW((long long)BL_ * D_), 256, 0, stream>>>(num, denom, attnb, (long long)BL_ * D_);
    gemm_bf16_wmma<128><<<dim3(BL_ / 128, D_ / 128, 1), 256, 0, stream>>>(
        attnb, woT, c5, D_, D_, D_, D_, 1, 0, 0, 0, 0, 0, 0);

    // ---- 8. residual + LN1 ----
    k_res_ln<<<BL_, 256, 0, stream>>>(x, c5, nullptr, ln1g, ln1b, out1, out1b);

    // ---- 9. FFN ----
    gemm_bf16_wmma<128><<<dim3(BL_ / 128, DFF_ / 128, 1), 256, 0, stream>>>(
        out1b, w1T, c6, D_, D_, D_, DFF_, 1, 0, 0, 0, 0, 0, 0);
    k_bias_elu<<<EW((long long)BL_ * DFF_), 256, 0, stream>>>(
        c6, b1, hb, (long long)BL_ * DFF_);
    gemm_bf16_wmma<128><<<dim3(BL_ / 128, D_ / 128, 1), 256, 0, stream>>>(
        hb, w2T, c7, DFF_, DFF_, DFF_, D_, 1, 0, 0, 0, 0, 0, 0);

    // ---- 10. residual + bias + LN2 -> d_out ----
    k_res_ln<<<BL_, 256, 0, stream>>>(out1, c7, b2, ln2g, ln2b, out, nullptr);
}
